// DeepRedModel_8589934783
// MI455X (gfx1250) — compile-verified, matrix operand order checked by reading, not run on hardware
//
#include <hip/hip_runtime.h>
#include <hip/hip_bf16.h>

// DeepRed co-attention forward for MI455X (gfx1250).
// One workgroup per batch element; both 200x128 embedding tiles staged in LDS
// as bf16; align GEMM done with v_wmma_f32_16x16x32_bf16 (f32 accumulate).

typedef __attribute__((ext_vector_type(16))) __bf16 v16bf;
typedef __attribute__((ext_vector_type(8)))  float  v8f;

#define B_SZ   1024
#define S_LEN  200
#define S_PAD  208            // 13 * 16
#define D_DIM  128
#define NT     13             // 16-wide tiles per side
#define NTILES (NT * NT)
#define ROW_U  (D_DIM / 2)    // 64 uints per row (bf16 pairs)

union FragBF { v16bf v; uint4 q[2]; };
union AccF   { v8f   v; float f[8]; };

__device__ __forceinline__ unsigned int pack_bf16x2(float lo, float hi) {
    unsigned int ul = __float_as_uint(lo);
    unsigned int uh = __float_as_uint(hi);
    ul = (ul + 0x7FFFu + ((ul >> 16) & 1u)) >> 16;   // round-to-nearest-even
    uh = (uh + 0x7FFFu + ((uh >> 16) & 1u)) >> 16;
    return (ul & 0xFFFFu) | (uh << 16);
}

__device__ __forceinline__ float bf16_bits_to_f(unsigned short u) {
    return __uint_as_float(((unsigned int)u) << 16);
}

// Hardware tanh on gfx1250 (v_tanh_f32); branch-free fallback otherwise.
__device__ __forceinline__ float fast_tanh(float x) {
#if __has_builtin(__builtin_amdgcn_tanhf)
    return __builtin_amdgcn_tanhf(x);
#else
    return fmaf(-2.f, __builtin_amdgcn_rcpf(__expf(2.f * x) + 1.f), 1.f);
#endif
}

// Load one 16x32 bf16 WMMA operand from LDS (A layout; B uses the same
// per-lane pattern since B[k,n] = E[n,k]).
// ISA 16-bit A 16x32 layout: lanes 0-15 -> M=lane, K = k0+{0..7, 16..23};
// lanes 16-31 -> M=lane-16, K = k0+{8..15, 24..31}.
__device__ __forceinline__ v16bf load_frag(const unsigned int* lds, int tile,
                                           int k0, int lane) {
    int r = tile * 16 + (lane & 15);
    int h = (lane >> 4) & 1;
    const unsigned int* base = lds + r * ROW_U + (k0 >> 1) + h * 4;
    FragBF f;
    f.q[0] = *reinterpret_cast<const uint4*>(base);       // K = k0 + h*8 .. +7
    f.q[1] = *reinterpret_cast<const uint4*>(base + 8);   // K = k0 + 16 + h*8 .. +7
    return f.v;
}

__global__ __launch_bounds__(256)
void deepred_coattn_kernel(const int*   __restrict__ user_nh,
                           const float* __restrict__ user_mask,
                           const int*   __restrict__ item_nh,
                           const float* __restrict__ item_mask,
                           const float* __restrict__ emb,    // (V+1, 128) f32
                           float*       __restrict__ out)    // [2, B, 128]
{
    extern __shared__ __align__(16) unsigned int smem[];
    unsigned int* lu = smem;                               // [S_PAD][ROW_U]
    unsigned int* li = lu + S_PAD * ROW_U;                 // [S_PAD][ROW_U]
    float* su  = reinterpret_cast<float*>(li + S_PAD * ROW_U); // user coef sums
    float* si  = su + S_PAD;                                   // item coef sums
    float* red = si + S_PAD;                                   // [256] scratch
    float* rsc = red + 256;                                    // [512] rep partials
    int*  idxu = reinterpret_cast<int*>(rsc + 512);            // [S_PAD]
    int*  idxi = idxu + S_PAD;                                 // [S_PAD]

    const int b    = blockIdx.x;
    const int tid  = threadIdx.x;
    const int lane = tid & 31;   // wave32

    // Stage neighbor indices, zero coef accumulators.
    if (tid < S_PAD) {
        idxu[tid] = (tid < S_LEN) ? user_nh[b * S_LEN + tid] : 0;
        idxi[tid] = (tid < S_LEN) ? item_nh[b * S_LEN + tid] : 0;
        su[tid] = 0.f;
        si[tid] = 0.f;
    }
    __syncthreads();

    // Gather embeddings f32 -> bf16 into LDS via float4 (global_load_b128).
    // idx==0 (padding_idx) and padded rows (s >= 200) become zeros, so
    // tanh(0)=0 falls out of the means without any masking.
    for (int idx = tid; idx < S_PAD * (D_DIM / 4); idx += 256) {
        int row = idx >> 5;                 // D_DIM/4 == 32 float4 per row
        int c4  = idx & 31;
        int gu = idxu[row];
        int gi = idxi[row];
        uint2 pu = {0u, 0u}, pi = {0u, 0u};
        if (gu != 0) {
            float4 e = reinterpret_cast<const float4*>(emb + (size_t)gu * D_DIM)[c4];
            pu.x = pack_bf16x2(e.x, e.y);
            pu.y = pack_bf16x2(e.z, e.w);
        }
        if (gi != 0) {
            float4 e = reinterpret_cast<const float4*>(emb + (size_t)gi * D_DIM)[c4];
            pi.x = pack_bf16x2(e.x, e.y);
            pi.y = pack_bf16x2(e.z, e.w);
        }
        reinterpret_cast<uint2*>(lu)[idx] = pu;
        reinterpret_cast<uint2*>(li)[idx] = pi;
    }
    __syncthreads();

    // align = tanh(E_u @ E_i^T). Each wave owns a contiguous row-major chunk
    // of the 13x13 tile grid. All loop control is scalar (readfirstlane) so
    // EXEC stays all-1s through the WMMA region. Row sums accumulate in
    // registers across the tj sweep (flushed per row-block); column sums are
    // one fire-and-forget ds_add_f32 per lane per tile. B fragments are
    // software-pipelined (bnext in flight while WMMA consumes bcur).
    {
        const int w    = __builtin_amdgcn_readfirstlane(tid >> 5);  // SGPR wave id
        const int hof  = ((lane >> 4) & 1) * 8;   // row offset of this half-wave
        const int ncol = lane & 15;               // column within tile
        const int t0 = (NTILES * w) >> 3;
        const int t1 = (NTILES * (w + 1)) >> 3;
        int ti = t0 / NT;                         // scalar div, once
        int tj = t0 - ti * NT;
        int remaining = t1 - t0;
        while (remaining > 0) {
            v16bf afrag[4];
            #pragma unroll
            for (int k = 0; k < 4; ++k)
                afrag[k] = load_frag(lu, ti, k * 32, lane);
            float rowacc[8];
            #pragma unroll
            for (int v = 0; v < 8; ++v) rowacc[v] = 0.f;

            const int span = (NT - tj < remaining) ? (NT - tj) : remaining;
            for (int e = 0; e < span; ++e, ++tj) {
                // rotating B pipeline: load k+1 before WMMA k consumes k
                v16bf bcur = load_frag(li, tj, 0, lane);
                AccF c = {};
                #pragma unroll
                for (int k = 0; k < 4; ++k) {
                    v16bf bnext;
                    if (k < 3) bnext = load_frag(li, tj, (k + 1) * 32, lane);
                    c.v = __builtin_amdgcn_wmma_f32_16x16x32_bf16(
                        false, afrag[k], false, bcur, (short)0, c.v,
                        false, false);
                    if (k < 3) bcur = bnext;
                }
                float t0v = fast_tanh(c.f[0]);
                rowacc[0] += t0v;
                float colpart = t0v;
                #pragma unroll
                for (int v = 1; v < 8; ++v) {
                    float t = fast_tanh(c.f[v]);
                    rowacc[v] += t;
                    colpart += t;
                }
                atomicAdd(&si[tj * 16 + ncol], colpart);  // 2-way conflict, no stall
            }
            const int mbase = ti * 16 + hof;
            #pragma unroll
            for (int v = 0; v < 8; ++v)
                atomicAdd(&su[mbase + v], rowacc[v]);
            ++ti;
            tj = 0;
            remaining -= span;
        }
    }
    __syncthreads();

    // Softmax over S for both coefficient vectors (masks are additive).
    float cu_ = (tid < S_LEN) ? su[tid] * (1.f / S_LEN) + user_mask[b * S_LEN + tid]
                              : -3.0e38f;
    float ci_ = (tid < S_LEN) ? si[tid] * (1.f / S_LEN) + item_mask[b * S_LEN + tid]
                              : -3.0e38f;

    red[tid] = cu_; __syncthreads();
    for (int s = 128; s > 0; s >>= 1) {
        if (tid < s) red[tid] = fmaxf(red[tid], red[tid + s]);
        __syncthreads();
    }
    float mxu = red[0]; __syncthreads();

    red[tid] = ci_; __syncthreads();
    for (int s = 128; s > 0; s >>= 1) {
        if (tid < s) red[tid] = fmaxf(red[tid], red[tid + s]);
        __syncthreads();
    }
    float mxi = red[0]; __syncthreads();

    float eu = (tid < S_LEN) ? __expf(cu_ - mxu) : 0.f;
    float ei = (tid < S_LEN) ? __expf(ci_ - mxi) : 0.f;

    red[tid] = eu; __syncthreads();
    for (int s = 128; s > 0; s >>= 1) {
        if (tid < s) red[tid] += red[tid + s];
        __syncthreads();
    }
    float sum_u = red[0]; __syncthreads();

    red[tid] = ei; __syncthreads();
    for (int s = 128; s > 0; s >>= 1) {
        if (tid < s) red[tid] += red[tid + s];
        __syncthreads();
    }
    float sum_i = red[0]; __syncthreads();

    if (tid < S_PAD) {
        su[tid] = (tid < S_LEN) ? eu / sum_u : 0.f;
        si[tid] = (tid < S_LEN) ? ei / sum_i : 0.f;
    }
    __syncthreads();

    // reps: rep[d] = sum_s E[s,d] * w[s].
    // tid -> (mat = tid>>7, p = d-pair 0..63, shalf = s-range half). Each
    // thread does 100 ds_load_b32 (a bf16 pair); partials combined via LDS.
    {
        const int mat   = tid >> 7;          // 0 = user, 1 = item
        const int sub   = tid & 127;
        const int p     = sub & 63;          // uint column (two d's)
        const int shalf = sub >> 6;          // which half of the s range
        const unsigned int* src = (mat == 0) ? lu : li;
        const float*        wv  = (mat == 0) ? su : si;
        float a0 = 0.f, a1 = 0.f;
        const int sBeg = shalf * (S_LEN / 2);
        const int sEnd = sBeg + (S_LEN / 2);
        #pragma unroll 4
        for (int s = sBeg; s < sEnd; ++s) {
            unsigned int pk = src[s * ROW_U + p];
            float wt = wv[s];
            a0 = fmaf(bf16_bits_to_f((unsigned short)(pk & 0xFFFFu)), wt, a0);
            a1 = fmaf(bf16_bits_to_f((unsigned short)(pk >> 16)), wt, a1);
        }
        // rsc layout: [mat][p][d&1][shalf]
        rsc[(((mat * 64 + p) * 2 + 0) * 2) + shalf] = a0;
        rsc[(((mat * 64 + p) * 2 + 1) * 2) + shalf] = a1;
    }
    __syncthreads();
    {
        const int mat = tid >> 7;
        const int d   = tid & 127;
        const int p   = d >> 1;
        const int j   = d & 1;
        float val = rsc[(((mat * 64 + p) * 2 + j) * 2) + 0]
                  + rsc[(((mat * 64 + p) * 2 + j) * 2) + 1];
        out[mat * B_SZ * D_DIM + b * D_DIM + d] = val;
    }
}

extern "C" void kernel_launch(void* const* d_in, const int* in_sizes, int n_in,
                              void* d_out, int out_size, void* d_ws, size_t ws_size,
                              hipStream_t stream) {
    // Input order: users, user_nh, user_mask, items, item_nh, item_mask, emb_table.
    // (users/items are unused by the reference forward.)
    const int*   user_nh   = (const int*)d_in[1];
    const float* user_mask = (const float*)d_in[2];
    const int*   item_nh   = (const int*)d_in[4];
    const float* item_mask = (const float*)d_in[5];
    const float* emb       = (const float*)d_in[6];
    float* out = (float*)d_out;

    size_t shmem = (size_t)(2 * S_PAD * ROW_U) * sizeof(unsigned int)  // lu + li
                 + (size_t)(2 * S_PAD) * sizeof(float)                 // su + si
                 + 256 * sizeof(float)                                 // red
                 + 512 * sizeof(float)                                 // rsc
                 + (size_t)(2 * S_PAD) * sizeof(int);                  // idxu + idxi

    deepred_coattn_kernel<<<B_SZ, 256, shmem, stream>>>(
        user_nh, user_mask, item_nh, item_mask, emb, out);
}
